// DiscreteSender_47674136985930
// MI455X (gfx1250) — matrix operand order repaired and checked by reading, not compiled, not run on hardware
//
#include <hip/hip_runtime.h>
#include <hip/hip_bf16.h>

typedef unsigned short u16;
typedef unsigned int   u32;
typedef __bf16 v16bf __attribute__((ext_vector_type(16)));
typedef float  v8f   __attribute__((ext_vector_type(8)));

#define B_   4096
#define A_   6
#define V_   1024
#define L_   32
#define DIN_ 64
#define H_   512

// packed-weight workspace layout (offsets in bf16 halves)
#define WI_OFF   0
#define WI_SZ    (384 * 512)
#define WIH_OFF  (WI_OFF + WI_SZ)
#define WIH_SZ   (64 * 2048)
#define WHH_OFF  (WIH_OFF + WIH_SZ)
#define WHH_SZ   (512 * 2048)
#define WO_OFF   (WHH_OFF + WHH_SZ)
#define WO_SZ    (512 * 1024)
#define PK_TOTAL (WO_OFF + WO_SZ)          // 1,900,544 halves = 3.8 MB

// LDS tile strides (halves); K + 8 padding -> conflict-free b128 row reads
#define SH_E 392   // encoder e-tile: 32 x 384
#define SH_H 520   // hidden tile:    32 x 512
#define SH_M 72    // emb tile:       32 x 64

__device__ __forceinline__ u16 f2bf(float f) {
  u32 u = __builtin_bit_cast(u32, f);
  u += 0x7FFFu + ((u >> 16) & 1u);     // round-to-nearest-even
  return (u16)(u >> 16);
}

struct Frag32B { uint4 lo, hi; };

__device__ __forceinline__ v16bf make_frag(uint4 lo, uint4 hi) {
  Frag32B f{lo, hi};
  return __builtin_bit_cast(v16bf, f);
}

// A-fragment (16x32 bf16) from a row-major LDS tile (contiguous k order, same as packed B):
// lane ln: row M=ln, half=0 -> K {0..15}, half=1 -> K {16..31}.
__device__ __forceinline__ v16bf lds_afrag(const u16* tile, int stride, int row,
                                           int kt, int half) {
  const uint4* p = (const uint4*)(tile + row * stride + kt * 32 + half * 16);
  return make_frag(p[0], p[1]);        // 16 contiguous halves = 2 uint4
}

// B-fragment (32x16 bf16) from packed global weights: lane's 16 halves contiguous.
__device__ __forceinline__ v16bf glb_bfrag(const u16* __restrict__ pk, int blk, int lane) {
  const uint4* p = (const uint4*)(pk + ((size_t)blk * 32 + lane) * 16);
  return make_frag(p[0], p[1]);
}

__device__ __forceinline__ v8f wmma_bf16(v16bf a, v16bf b, v8f c) {
  return __builtin_amdgcn_wmma_f32_16x16x32_bf16(
      /*neg_a=*/false, a, /*neg_b=*/false, b, /*c_mod=*/(short)0, c,
      /*reuse_a=*/false, /*reuse_b=*/false);
}

__device__ __forceinline__ v8f zero8() {
  v8f z;
#pragma unroll
  for (int i = 0; i < 8; ++i) z[i] = 0.f;
  return z;
}

__device__ __forceinline__ float sigm(float x)  { return 1.f / (1.f + __expf(-x)); }
__device__ __forceinline__ float tanhx(float x) { return 2.f / (1.f + __expf(-2.f * x)) - 1.f; }

// ---------------- prep: fp32 -> blocked bf16 fragment layout ----------------
// Block (nt, kt) is a 32x16 (KxN) tile; dst[((nt*KT+kt)*32 + lane)*16 + s].
// B layout in contiguous k order, as lds_afrag uses (n across lanes; K split by lane half):
//   lane l: column n = nt*16 + (l&15);  k = kt*32 + s + 16*(l>=16), s = 0..15.
__global__ void pack_w_kernel(const float* __restrict__ src, u16* __restrict__ dst,
                              int K, int N, int trans) {
  const int KT = K >> 5;
  const int total = K * N;
  for (int i = blockIdx.x * blockDim.x + threadIdx.x; i < total;
       i += gridDim.x * blockDim.x) {
    int blk    = i >> 9;
    int within = i & 511;
    int lane   = within >> 4;
    int s      = within & 15;
    int nt = blk / KT, kt = blk - nt * KT;
    int k = kt * 32 + s + ((lane >> 4) << 4);
    int n = nt * 16 + (lane & 15);
    float v = trans ? src[(size_t)n * K + k] : src[(size_t)k * N + n];
    dst[i] = f2bf(v);
  }
}

__global__ void bias_sum_kernel(const float* __restrict__ bih,
                                const float* __restrict__ bhh,
                                float* __restrict__ out) {
  int i = blockIdx.x * blockDim.x + threadIdx.x;
  if (i < 4 * H_) out[i] = bih[i] + bhh[i];
}

// ---------------- fused persistent LSTM decoder ----------------
// 128 WGs x 1024 threads. WG owns 32 batch rows for the whole sequence.
// Wave w: rt = w>>4 (row tile), jw = w&15 (32-wide j-slice of H).
// Gate accumulators for a wave cover matching j columns of all four gate
// sections (nt = g*32 + 2*jw + t16), so the LSTM cell update is pure VALU and
// the cell state c lives in VGPRs for the entire 32-step sequence.
__global__ __launch_bounds__(1024) void lstm_decode_kernel(
    const int*   __restrict__ x,
    const float* __restrict__ input_emb,
    const float* __restrict__ output_emb,
    const float* __restrict__ bi,
    const float* __restrict__ bo,
    const float* __restrict__ sos,
    const u16*   __restrict__ wi_pk,
    const u16*   __restrict__ wih_pk,
    const u16*   __restrict__ whh_pk,
    const u16*   __restrict__ wo_pk,
    const float* __restrict__ bias_g,
    float* __restrict__ seq_out,
    float* __restrict__ logit_out) {
  __shared__ __align__(16) u16 sE[32 * SH_E];
  __shared__ __align__(16) u16 sH[32 * SH_H];
  __shared__ __align__(16) u16 sM[32 * SH_M];
  __shared__ float aVal[32][16];
  __shared__ int   aIdx[32][16];

  const int tid  = threadIdx.x;
  const int w    = tid >> 5;
  const int lane = tid & 31;
  const int half = lane >> 4;
  const int ln   = lane & 15;
  const int rt   = w >> 4;
  const int jw   = w & 15;
  const int b0   = blockIdx.x * 32;
  const int rowA = rt * 16;

  // ---- stage encoder input tile e = input_emb[x].reshape(32, 384) as bf16 ----
  for (int i = tid; i < 32 * 384; i += 1024) {
    int m = i / 384, col = i - m * 384;
    int a = col >> 6, d = col & 63;
    int idx = x[(b0 + m) * A_ + a];
    sE[m * SH_E + col] = f2bf(input_emb[(size_t)idx * DIN_ + d]);
  }
  // ---- emb0 = SOS ----
  for (int i = tid; i < 32 * 64; i += 1024) {
    int m = i >> 6, col = i & 63;
    sM[m * SH_M + col] = f2bf(sos[col]);
  }
  __syncthreads();

  // ---- h0 = e @ Wi + bi   (K=384 -> 12 k-tiles; wave cols jw*32..+31) ----
  v8f h0[2] = {zero8(), zero8()};
  for (int kt = 0; kt < 12; ++kt) {
    v16bf a = lds_afrag(sE, SH_E, rowA + ln, kt, half);
    v16bf bf0 = glb_bfrag(wi_pk, (2 * jw + 0) * 12 + kt, lane);
    v16bf bf1 = glb_bfrag(wi_pk, (2 * jw + 1) * 12 + kt, lane);
    h0[0] = wmma_bf16(a, bf0, h0[0]);
    h0[1] = wmma_bf16(a, bf1, h0[1]);
  }
  float cst[2][8];
#pragma unroll
  for (int t16 = 0; t16 < 2; ++t16) {
    float bv = bi[jw * 32 + t16 * 16 + ln];
#pragma unroll
    for (int r = 0; r < 8; ++r) {
      h0[t16][r] += bv;
      cst[t16][r] = 0.f;
      sH[(rowA + r + 8 * half) * SH_H + jw * 32 + t16 * 16 + ln] = f2bf(h0[t16][r]);
    }
  }

  // ---- 32 decode steps, fully fused ----
  for (int t = 0; t < L_; ++t) {
    __syncthreads();  // h / emb of this step visible to all waves

    // gates[16 x 2048]: wave owns j-slice in all 4 gate sections.
    v8f acc[4][2];
#pragma unroll
    for (int g = 0; g < 4; ++g)
#pragma unroll
      for (int t16 = 0; t16 < 2; ++t16) acc[g][t16] = zero8();

    // emb @ W_ih^T  (K=64): batch all 8 B-frag loads, then consume in order
    // so load latency hides behind earlier WMMAs (staged s_wait_loadcnt).
#pragma unroll
    for (int kt = 0; kt < 2; ++kt) {
      v16bf a = lds_afrag(sM, SH_M, rowA + ln, kt, half);
      v16bf bfr[8];
#pragma unroll
      for (int u = 0; u < 8; ++u) {
        int nt = (u >> 1) * 32 + 2 * jw + (u & 1);
        bfr[u] = glb_bfrag(wih_pk, nt * 2 + kt, lane);
      }
#pragma unroll
      for (int u = 0; u < 8; ++u)
        acc[u >> 1][u & 1] = wmma_bf16(a, bfr[u], acc[u >> 1][u & 1]);
    }
    // h @ W_hh^T  (K=512), same load-batching per k-tile.
    for (int kt = 0; kt < 16; ++kt) {
      v16bf a = lds_afrag(sH, SH_H, rowA + ln, kt, half);
      v16bf bfr[8];
#pragma unroll
      for (int u = 0; u < 8; ++u) {
        int nt = (u >> 1) * 32 + 2 * jw + (u & 1);
        bfr[u] = glb_bfrag(whh_pk, nt * 16 + kt, lane);
      }
#pragma unroll
      for (int u = 0; u < 8; ++u)
        acc[u >> 1][u & 1] = wmma_bf16(a, bfr[u], acc[u >> 1][u & 1]);
    }

    // LSTM cell update: i/f/g/o for (row, j) share lane/slot -> pure VALU.
    float hn[2][8];
#pragma unroll
    for (int t16 = 0; t16 < 2; ++t16) {
      int nb = jw * 32 + t16 * 16 + ln;
      float bI = bias_g[nb];
      float bF = bias_g[512 + nb];
      float bG = bias_g[1024 + nb];
      float bO = bias_g[1536 + nb];
#pragma unroll
      for (int r = 0; r < 8; ++r) {
        float iv = sigm(acc[0][t16][r] + bI);
        float fv = sigm(acc[1][t16][r] + bF);
        float gv = tanhx(acc[2][t16][r] + bG);
        float ov = sigm(acc[3][t16][r] + bO);
        float cv = fv * cst[t16][r] + iv * gv;
        cst[t16][r] = cv;
        hn[t16][r]  = ov * tanhx(cv);
      }
    }
    __syncthreads();  // everyone done reading old h
#pragma unroll
    for (int t16 = 0; t16 < 2; ++t16)
#pragma unroll
      for (int r = 0; r < 8; ++r)
        sH[(rowA + r + 8 * half) * SH_H + jw * 32 + t16 * 16 + ln] = f2bf(hn[t16][r]);
    __syncthreads();  // new h visible

    // logits = h @ Wo + bo  (wave owns cols jw*64..+63)
    v8f lacc[4];
#pragma unroll
    for (int n4 = 0; n4 < 4; ++n4) lacc[n4] = zero8();
#pragma unroll 2
    for (int kt = 0; kt < 16; ++kt) {
      v16bf a = lds_afrag(sH, SH_H, rowA + ln, kt, half);
      v16bf bfr[4];
#pragma unroll
      for (int n4 = 0; n4 < 4; ++n4)
        bfr[n4] = glb_bfrag(wo_pk, (4 * jw + n4) * 16 + kt, lane);
#pragma unroll
      for (int n4 = 0; n4 < 4; ++n4)
        lacc[n4] = wmma_bf16(a, bfr[n4], lacc[n4]);
    }
#pragma unroll
    for (int n4 = 0; n4 < 4; ++n4) {
      float bv = bo[jw * 64 + n4 * 16 + ln];
#pragma unroll
      for (int r = 0; r < 8; ++r) {
        float v = lacc[n4][r] + bv;
        lacc[n4][r] = v;
        int row = rowA + r + 8 * half;
        // streaming 537 MB output: non-temporal so weights stay in L2
        __builtin_nontemporal_store(
            v, &logit_out[((size_t)(b0 + row) * L_ + t) * V_ + jw * 64 + n4 * 16 + ln]);
      }
    }

    // per-row argmax: lane-local over 4 tiles, then shfl over 16-lane half.
#pragma unroll
    for (int r = 0; r < 8; ++r) {
      float bvv = lacc[0][r];
      int   bii = jw * 64 + ln;
#pragma unroll
      for (int n4 = 1; n4 < 4; ++n4) {
        float cv = lacc[n4][r];
        int   ci = jw * 64 + n4 * 16 + ln;
        if (cv > bvv || (cv == bvv && ci < bii)) { bvv = cv; bii = ci; }
      }
#pragma unroll
      for (int m = 1; m <= 8; m <<= 1) {
        float ov = __shfl_xor(bvv, m, 32);
        int   oi = __shfl_xor(bii, m, 32);
        if (ov > bvv || (ov == bvv && oi < bii)) { bvv = ov; bii = oi; }
      }
      if (ln == 0) {
        int row = rowA + r + 8 * half;
        aVal[row][jw] = bvv;
        aIdx[row][jw] = bii;
      }
    }
    __syncthreads();

    // cross-wave reduce: wave w finishes row w, writes sym, refills emb row.
    float rv = -3.4e38f;
    int   ri = 0;
    if (lane < 16) { rv = aVal[w][lane]; ri = aIdx[w][lane]; }
#pragma unroll
    for (int m = 1; m <= 8; m <<= 1) {
      float ov = __shfl_xor(rv, m, 32);
      int   oi = __shfl_xor(ri, m, 32);
      if (ov > rv || (ov == rv && oi < ri)) { rv = ov; ri = oi; }
    }
    int sym = __shfl(ri, 0, 32);
    if (lane == 0) seq_out[(size_t)(b0 + w) * L_ + t] = (float)sym;
    sM[w * SH_M + lane]      = f2bf(output_emb[(size_t)sym * DIN_ + lane]);
    sM[w * SH_M + lane + 32] = f2bf(output_emb[(size_t)sym * DIN_ + lane + 32]);
  }
}

extern "C" void kernel_launch(void* const* d_in, const int* in_sizes, int n_in,
                              void* d_out, int out_size, void* d_ws, size_t ws_size,
                              hipStream_t stream) {
  const int*   x          = (const int*)  d_in[0];
  const float* input_emb  = (const float*)d_in[1];
  const float* output_emb = (const float*)d_in[2];
  const float* Wi         = (const float*)d_in[3];
  const float* bi         = (const float*)d_in[4];
  const float* W_ih       = (const float*)d_in[5];
  const float* W_hh       = (const float*)d_in[6];
  const float* b_ih       = (const float*)d_in[7];
  const float* b_hh       = (const float*)d_in[8];
  const float* Wo         = (const float*)d_in[9];
  const float* bo         = (const float*)d_in[10];
  const float* sos        = (const float*)d_in[11];

  u16*   wsh    = (u16*)d_ws;
  u16*   wi_pk  = wsh + WI_OFF;
  u16*   wih_pk = wsh + WIH_OFF;
  u16*   whh_pk = wsh + WHH_OFF;
  u16*   wo_pk  = wsh + WO_OFF;
  float* bias_g = (float*)((char*)d_ws + (size_t)PK_TOTAL * 2);

  // pack weights fp32 -> blocked bf16 fragments (L2-resident, ~3.8 MB)
  pack_w_kernel<<<256, 256, 0, stream>>>(Wi,   wi_pk,  384, 512,  0);
  pack_w_kernel<<<256, 256, 0, stream>>>(W_ih, wih_pk, 64,  2048, 1);
  pack_w_kernel<<<256, 256, 0, stream>>>(W_hh, whh_pk, 512, 2048, 1);
  pack_w_kernel<<<256, 256, 0, stream>>>(Wo,   wo_pk,  512, 1024, 0);
  bias_sum_kernel<<<8, 256, 0, stream>>>(b_ih, b_hh, bias_g);

  float* seq_out   = (float*)d_out;
  float* logit_out = seq_out + (size_t)B_ * L_;

  lstm_decode_kernel<<<B_ / 32, 1024, 0, stream>>>(
      x, input_emb, output_emb, bi, bo, sos,
      wi_pk, wih_pk, whh_pk, wo_pk, bias_g, seq_out, logit_out);
}
